// Global_GNN_74285754351875
// MI455X (gfx1250) — compile-verified, matrix operand order checked by reading, not run on hardware
//
#include <hip/hip_runtime.h>

// ---------------------------------------------------------------------------
// CDNA5 (gfx1250) ChebConv GNN forward.
// Dense transforms: v_wmma_f32_16x16x32_f16 (wave32), B pre-swizzled in LDS so
// each operand is one 32B ds load; A via global_load_b128 + f16 pack.
// Edge propagation: float4 gather + global_atomic_add_f32 scatter (HBM-bound).
// ---------------------------------------------------------------------------

typedef __attribute__((ext_vector_type(16))) _Float16 v16h;
typedef __attribute__((ext_vector_type(8)))  float    v8f;

#define BN_EPS 1e-5f
#define NG 64          // graphs
#define C  20          // hidden channels

// ISA 7.12.2: 16-bit A/B operand K-mapping for 16x16x32 WMMA.
// lanes 0-15 hold K halves {0..7, 16..23}; lanes 16-31 hold {8..15, 24..31}.
__device__ __forceinline__ int wmma_kmap(int lane, int e) {
  return (e & 7) + ((lane & 16) ? 8 : 0) + ((e & 8) ? 16 : 0);
}

// U[N][64] = H[N][K] @ [W0|W1|W2] (each K x 20 -> K x 60, padded to Kpad x 64).
// WIDE: layer 0 (K=140, Kpad=160). !WIDE: layers 1..3 (K=20, Kpad=32).
// One wave computes a 16-row x 64-col tile; B staged in LDS in WMMA register
// order so each v16h operand is a single aligned 32-byte LDS load.
template <bool WIDE>
__global__ void cheb_gemm_wmma(const float* __restrict__ H, int N,
                               const float* __restrict__ W,   // [3][K][20]
                               float* __restrict__ U) {       // [N][64]
  constexpr int K   = WIDE ? 140 : 20;
  constexpr int NKC = WIDE ? 5 : 1;          // K-chunks of 32
  __shared__ __align__(32) _Float16 wlds[NKC * 4 * 32 * 16];

  const int tid = threadIdx.x;
  // Stage W into LDS pre-swizzled to WMMA B-operand register layout:
  // wlds[((kc*4 + nt)*32 + lane)*16 + e] = W[k = kc*32+kmap(lane,e)][n = nt*16+(lane&15)]
  for (int idx = tid; idx < NKC * 4 * 32 * 16; idx += blockDim.x) {
    const int e    = idx & 15;
    const int lane = (idx >> 4) & 31;
    const int nt   = (idx >> 9) & 3;
    const int kc   = idx >> 11;
    const int k    = kc * 32 + wmma_kmap(lane, e);
    const int n    = nt * 16 + (lane & 15);
    float v = 0.f;
    if (k < K && n < 60) v = W[((n / 20) * K + k) * 20 + (n % 20)];
    wlds[idx] = (_Float16)v;
  }
  __syncthreads();

  const int lane = tid & 31;
  const int rowTile = blockIdx.x * (blockDim.x >> 5) + (tid >> 5);
  const int row0 = rowTile * 16;
  const int m = lane & 15;
  const int arow = row0 + m;
  const bool hi = (lane & 16) != 0;
  const float* __restrict__ hrow = H + (long)arow * K;

  v8f acc0 = {}, acc1 = {}, acc2 = {}, acc3 = {};
#pragma unroll
  for (int kc = 0; kc < NKC; ++kc) {
    v16h a;
    if (WIDE) {
      const int base = kc * 32 + (hi ? 8 : 0);
      if (arow < N && base + 24 <= K) {
        // Two contiguous 8-float runs: base..base+7 and base+16..base+23.
        const float4 x0 = *(const float4*)(hrow + base);
        const float4 x1 = *(const float4*)(hrow + base + 4);
        const float4 y0 = *(const float4*)(hrow + base + 16);
        const float4 y1 = *(const float4*)(hrow + base + 20);
        a[0] = (_Float16)x0.x; a[1] = (_Float16)x0.y; a[2] = (_Float16)x0.z; a[3] = (_Float16)x0.w;
        a[4] = (_Float16)x1.x; a[5] = (_Float16)x1.y; a[6] = (_Float16)x1.z; a[7] = (_Float16)x1.w;
        a[8] = (_Float16)y0.x; a[9] = (_Float16)y0.y; a[10] = (_Float16)y0.z; a[11] = (_Float16)y0.w;
        a[12] = (_Float16)y1.x; a[13] = (_Float16)y1.y; a[14] = (_Float16)y1.z; a[15] = (_Float16)y1.w;
      } else {  // K tail chunk / OOB rows
#pragma unroll
        for (int e = 0; e < 16; ++e) {
          const int k = base + (e & 7) + ((e & 8) ? 16 : 0);
          const float v = (arow < N && k < K) ? hrow[k] : 0.f;
          a[e] = (_Float16)v;
        }
      }
    } else {
      // K=20: load the whole row (5 float4s, always in bounds), select per group.
      float r[20];
      if (arow < N) {
        const float4 t0 = *(const float4*)(hrow);
        const float4 t1 = *(const float4*)(hrow + 4);
        const float4 t2 = *(const float4*)(hrow + 8);
        const float4 t3 = *(const float4*)(hrow + 12);
        const float4 t4 = *(const float4*)(hrow + 16);
        r[0] = t0.x; r[1] = t0.y; r[2] = t0.z; r[3] = t0.w;
        r[4] = t1.x; r[5] = t1.y; r[6] = t1.z; r[7] = t1.w;
        r[8] = t2.x; r[9] = t2.y; r[10] = t2.z; r[11] = t2.w;
        r[12] = t3.x; r[13] = t3.y; r[14] = t3.z; r[15] = t3.w;
        r[16] = t4.x; r[17] = t4.y; r[18] = t4.z; r[19] = t4.w;
      } else {
#pragma unroll
        for (int i = 0; i < 20; ++i) r[i] = 0.f;
      }
      if (!hi) {  // K halves {0..7, 16..23}; 20..23 are pad
#pragma unroll
        for (int e = 0; e < 8; ++e) a[e] = (_Float16)r[e];
#pragma unroll
        for (int e = 8; e < 12; ++e) a[e] = (_Float16)r[e + 8];
#pragma unroll
        for (int e = 12; e < 16; ++e) a[e] = (_Float16)0.f;
      } else {    // K halves {8..15, 24..31}; 24..31 are pad
#pragma unroll
        for (int e = 0; e < 8; ++e) a[e] = (_Float16)r[e + 8];
#pragma unroll
        for (int e = 8; e < 16; ++e) a[e] = (_Float16)0.f;
      }
    }

    const _Float16* bb = &wlds[((kc * 4) * 32 + lane) * 16];
    const v16h b0 = *(const v16h*)(bb);
    const v16h b1 = *(const v16h*)(bb + 512);
    const v16h b2 = *(const v16h*)(bb + 1024);
    const v16h b3 = *(const v16h*)(bb + 1536);
    acc0 = __builtin_amdgcn_wmma_f32_16x16x32_f16(false, a, false, b0, (short)0, acc0, false, false);
    acc1 = __builtin_amdgcn_wmma_f32_16x16x32_f16(false, a, false, b1, (short)0, acc1, false, false);
    acc2 = __builtin_amdgcn_wmma_f32_16x16x32_f16(false, a, false, b2, (short)0, acc2, false, false);
    acc3 = __builtin_amdgcn_wmma_f32_16x16x32_f16(false, a, false, b3, (short)0, acc3, false, false);
  }

  // D layout: M = r + 8*(lane>=16), Ncol = (lane&15) + 16*ntile.
#pragma unroll
  for (int r = 0; r < 8; ++r) {
    const int orow = row0 + r + (hi ? 8 : 0);
    if (orow < N) {
      float* out = U + (long)orow * 64 + m;
      out[0]  = acc0[r];
      out[16] = acc1[r];
      out[32] = acc2[r];
      if (m < 12) out[48] = acc3[r];   // cols 48..59 only
    }
  }
}

__global__ void deg_kernel(const int* __restrict__ src, float* __restrict__ deg, int E) {
  const int e = blockIdx.x * blockDim.x + threadIdx.x;
  if (e < E) atomicAdd(&deg[src[e]], 1.0f);
}

__global__ void dis_kernel(float* __restrict__ dis, int N) {
  const int n = blockIdx.x * blockDim.x + threadIdx.x;
  if (n < N) {
    const float d = dis[n];
    dis[n] = (d > 0.f) ? rsqrtf(fmaxf(d, 1.0f)) : 0.f;
  }
}

// out[dst][c] += (-dis[src]*dis[dst]) * in[src*stride + c], c in [0,20)
__global__ void prop_kernel(const float* __restrict__ in, int in_stride,
                            float* __restrict__ out,
                            const float* __restrict__ dis,
                            const int* __restrict__ src,
                            const int* __restrict__ dst, int E) {
  const int e = blockIdx.x * blockDim.x + threadIdx.x;
  if (e >= E) return;
  const int s = src[e], d = dst[e];
  const float w = -dis[s] * dis[d];
  const float* __restrict__ hin = in + (long)s * in_stride;
  const float4 h0 = *(const float4*)(hin);
  const float4 h1 = *(const float4*)(hin + 4);
  const float4 h2 = *(const float4*)(hin + 8);
  const float4 h3 = *(const float4*)(hin + 12);
  const float4 h4 = *(const float4*)(hin + 16);
  const float hl[20] = {h0.x, h0.y, h0.z, h0.w, h1.x, h1.y, h1.z, h1.w,
                        h2.x, h2.y, h2.z, h2.w, h3.x, h3.y, h3.z, h3.w,
                        h4.x, h4.y, h4.z, h4.w};
  float* po = out + (long)d * C;
#pragma unroll
  for (int c = 0; c < C; ++c) atomicAdd(po + c, w * hl[c]);
}

// p <- u1 + 2*p  (u1 = U[:,20:40])
__global__ void fuse_kernel(const float* __restrict__ U, float* __restrict__ p, int N) {
  const int idx = blockIdx.x * blockDim.x + threadIdx.x;
  if (idx >= N * C) return;
  const int row = idx / C, c = idx - row * C;
  p[idx] = U[(long)row * 64 + 20 + c] + 2.0f * p[idx];
}

// out = relu(bn(u0 - u2 + q + bias)) [+ 0.7*prev]
__global__ void post_kernel(const float* __restrict__ U, const float* __restrict__ q,
                            const float* __restrict__ bias,
                            const float* __restrict__ gamma, const float* __restrict__ beta,
                            const float* __restrict__ mean,  const float* __restrict__ var,
                            const float* __restrict__ prev, float resw,
                            float* __restrict__ out, int N) {
  const int idx = blockIdx.x * blockDim.x + threadIdx.x;
  if (idx >= N * C) return;
  const int row = idx / C, c = idx - row * C;
  float z = U[(long)row * 64 + c] - U[(long)row * 64 + 40 + c] + q[idx] + bias[c];
  z = (z - mean[c]) * rsqrtf(var[c] + BN_EPS) * gamma[c] + beta[c];
  z = fmaxf(z, 0.f);
  if (prev) z += resw * prev[idx];
  out[idx] = z;
}

// Jump-knowledge softmax-weighted sum + two-level (LDS then global) mean-pool.
__global__ void jk_pool_kernel(const float* __restrict__ lo,   // [4][N][20]
                               const float* __restrict__ lw,   // [4]
                               const int* __restrict__ batch,  // [N]
                               float* __restrict__ pooled,     // [64][20]
                               float* __restrict__ cnt, int N) {
  __shared__ float sp[NG * C];
  __shared__ float sc[NG];
  for (int i = threadIdx.x; i < NG * C; i += blockDim.x) sp[i] = 0.f;
  for (int i = threadIdx.x; i < NG; i += blockDim.x) sc[i] = 0.f;
  __syncthreads();

  const int n = blockIdx.x * blockDim.x + threadIdx.x;
  if (n < N) {
    const float l0 = lw[0], l1 = lw[1], l2 = lw[2], l3 = lw[3];
    const float mx = fmaxf(fmaxf(l0, l1), fmaxf(l2, l3));
    float w0 = __expf(l0 - mx), w1 = __expf(l1 - mx), w2 = __expf(l2 - mx), w3 = __expf(l3 - mx);
    const float inv = 1.0f / (w0 + w1 + w2 + w3);
    w0 *= inv; w1 *= inv; w2 *= inv; w3 *= inv;
    const int g = batch[n];
    const long base = (long)n * C, stride = (long)N * C;
    float* sg = sp + g * C;
#pragma unroll
    for (int c = 0; c < C; ++c) {
      const float emb = w0 * lo[base + c] + w1 * lo[stride + base + c] +
                        w2 * lo[2 * stride + base + c] + w3 * lo[3 * stride + base + c];
      atomicAdd(sg + c, emb);       // ds_add_f32 (LDS)
    }
    atomicAdd(&sc[g], 1.0f);
  }
  __syncthreads();

  // batch is sorted -> each block touches ~1 graph; flush only non-zeros.
  for (int i = threadIdx.x; i < NG * C; i += blockDim.x)
    if (sp[i] != 0.f) atomicAdd(&pooled[i], sp[i]);
  for (int i = threadIdx.x; i < NG; i += blockDim.x)
    if (sc[i] != 0.f) atomicAdd(&cnt[i], sc[i]);
}

__global__ void fc_kernel(const float* __restrict__ pooled, const float* __restrict__ cnt,
                          const float* __restrict__ fcW,  // [20][2]
                          const float* __restrict__ fcb,  // [2]
                          float* __restrict__ out) {      // [64][2]
  const int g = threadIdx.x;
  if (g >= NG) return;
  const float invc = 1.0f / fmaxf(cnt[g], 1.0f);
#pragma unroll
  for (int o = 0; o < 2; ++o) {
    float acc = fcb[o];
#pragma unroll
    for (int c = 0; c < C; ++c) acc += pooled[g * C + c] * invc * fcW[c * 2 + o];
    out[g * 2 + o] = acc;
  }
}

extern "C" void kernel_launch(void* const* d_in, const int* in_sizes, int n_in,
                              void* d_out, int out_size, void* d_ws, size_t ws_size,
                              hipStream_t stream) {
  const float* x       = (const float*)d_in[0];    // [N,140]
  const float* convW0  = (const float*)d_in[1];    // [3,140,20]
  const float* convb0  = (const float*)d_in[2];    // [20]
  const float* convW   = (const float*)d_in[3];    // [3,3,20,20]
  const float* convb   = (const float*)d_in[4];    // [3,20]
  const float* bn_g    = (const float*)d_in[5];    // [4,20]
  const float* bn_b    = (const float*)d_in[6];
  const float* bn_m    = (const float*)d_in[7];
  const float* bn_v    = (const float*)d_in[8];
  const float* lw      = (const float*)d_in[9];    // [4]
  const float* fcW     = (const float*)d_in[10];   // [20,2]
  const float* fcb     = (const float*)d_in[11];   // [2]
  const int*   ei      = (const int*)d_in[12];     // [2,E] int32 (x64 disabled)
  const int*   batch   = (const int*)d_in[13];     // [N]

  const int N = in_sizes[0] / 140;
  const int E = in_sizes[12] / 2;
  const int* src = ei;
  const int* dst = ei + E;

  // Workspace layout (floats).
  float* ws     = (float*)d_ws;
  float* dis    = ws;                         // N
  float* U      = dis + N;                    // N*64
  float* p      = U + (size_t)N * 64;         // N*20
  float* q      = p + (size_t)N * C;          // N*20
  float* lo     = q + (size_t)N * C;          // 4*N*20
  float* pooled = lo + (size_t)4 * N * C;     // 64*20
  float* cnt    = pooled + NG * C;            // 64

  const int TB = 256;
  const int gE  = (E + TB - 1) / TB;
  const int gNC = (N * C + TB - 1) / TB;
  const int gN  = (N + TB - 1) / TB;
  const int gMM = (N + 127) / 128;            // 8 waves/block * 16 rows

  // Normalization: deg -> dis (in place).
  hipMemsetAsync(dis, 0, (size_t)N * sizeof(float), stream);
  deg_kernel<<<gE, TB, 0, stream>>>(src, dis, E);
  dis_kernel<<<gN, TB, 0, stream>>>(dis, N);

  for (int L = 0; L < 4; ++L) {
    const float* bl = (L == 0) ? convb0 : convb + (size_t)(L - 1) * C;

    // u0|u1|u2 = H @ [W0|W1|W2]   (WMMA)
    if (L == 0) {
      cheb_gemm_wmma<true><<<gMM, TB, 0, stream>>>(x, N, convW0, U);
    } else {
      cheb_gemm_wmma<false><<<gMM, TB, 0, stream>>>(lo + (size_t)(L - 1) * N * C, N,
                                                    convW + (size_t)(L - 1) * 3 * C * C, U);
    }

    // p = S*u2 ; p = u1 + 2p ; q = S*p ; out = u0 - u2 + q (+bias,BN,ReLU,res)
    hipMemsetAsync(p, 0, (size_t)N * C * sizeof(float), stream);
    prop_kernel<<<gE, TB, 0, stream>>>(U + 40, 64, p, dis, src, dst, E);
    fuse_kernel<<<gNC, TB, 0, stream>>>(U, p, N);
    hipMemsetAsync(q, 0, (size_t)N * C * sizeof(float), stream);
    prop_kernel<<<gE, TB, 0, stream>>>(p, C, q, dis, src, dst, E);

    const float* prev = (L == 0) ? nullptr : lo + (size_t)(L - 1) * N * C;
    post_kernel<<<gNC, TB, 0, stream>>>(U, q, bl,
                                        bn_g + L * C, bn_b + L * C,
                                        bn_m + L * C, bn_v + L * C,
                                        prev, 0.7f,
                                        lo + (size_t)L * N * C, N);
  }

  // Jump knowledge + mean pool + FC.
  hipMemsetAsync(pooled, 0, (size_t)(NG * C + NG) * sizeof(float), stream);
  jk_pool_kernel<<<gN, TB, 0, stream>>>(lo, lw, batch, pooled, cnt, N);
  fc_kernel<<<1, 64, 0, stream>>>(pooled, cnt, fcW, fcb, (float*)d_out);
}
